// NNConvModel_57294863729414
// MI455X (gfx1250) — matrix operand order repaired and checked by reading, not compiled, not run on hardware
//
#include <hip/hip_runtime.h>

// ---------------------------------------------------------------------------
// NNConv GNN for MI455X (gfx1250, wave32, WMMA).
//
// Heavy math: per-edge  msg = (hidden ⊗ h[src]) @ reshape(e2W,[1024,32])
// implemented as f16 WMMA 16x16x32 GEMM with K=1024 (+1 bias step folding e2b),
// B matrix staged once per workgroup in LDS in fragment-native layout.
// Inner loop double-buffers both the A fragment (avoids the 4-coexec
// WMMA->VALU WAR hazard nops) and the B fragments (hides ds_load latency).
// Everything else (segment means, GraphNorm, pooling, MLP head) is
// bandwidth-trivial and uses plain VALU + f32 global atomics.
// ---------------------------------------------------------------------------

#define N_NODES  65536
#define N_EDGES  131072
#define N_GRAPHS 2048
#define EPSV     1e-5f

typedef __attribute__((ext_vector_type(16))) _Float16 v16h;
typedef __attribute__((ext_vector_type(8)))  float    v8f;

// --------------------------- utility kernels -------------------------------

__global__ void zero_kernel(float* __restrict__ p, int n) {
  int i = blockIdx.x * blockDim.x + threadIdx.x;
  if (i < n) p[i] = 0.0f;
}

__global__ void deg_kernel(const int* __restrict__ ei, float* __restrict__ deg) {
  int e = blockIdx.x * blockDim.x + threadIdx.x;
  if (e < N_EDGES) atomicAdd(deg + (unsigned)ei[N_EDGES + e], 1.0f);
}

__global__ void gcnt_kernel(const int* __restrict__ batch, float* __restrict__ gcnt) {
  int n = blockIdx.x * blockDim.x + threadIdx.x;
  if (n < N_NODES) atomicAdd(gcnt + (unsigned)batch[n], 1.0f);
}

// --------------------------- edge WMMA kernel ------------------------------
// LDS layout (dynamic):
//   [0, 67584)        : B fragments, f16.  33 k-steps x 2 n-tiles x 32 lanes x 16 halfs
//   [67584, 76288)    : per-wave hidden scalars, 4 waves x 16 rows x 34 f32
//   [76288, 76544)    : per-wave dst indices,   4 waves x 16 i32
//   [76544, 77568)    : e1W layer slice (8x32 f32)
//   [77568, 77696)    : e1b layer slice (32 f32)
#define SMEM_B_HALFS (33 * 2 * 32 * 16)
#define OFF_HID 67584
#define OFF_DST 76288
#define OFF_E1W 76544
#define OFF_E1B 77568
#define SMEM_TOTAL 77696

__global__ __launch_bounds__(128) void edge_wmma_kernel(
    const float* __restrict__ hsrc, const float* __restrict__ edge_attr,
    const float* __restrict__ e1W, const float* __restrict__ e1b,
    const float* __restrict__ e2W, const float* __restrict__ e2b,
    const int* __restrict__ edge_index, float* __restrict__ agg, int layer)
{
  extern __shared__ v16h smemv[];        // v16h decl => 32B base alignment
  char* smem = (char*)smemv;
  _Float16* Bl = (_Float16*)smem;

  const int tid  = threadIdx.x;
  const int wave = tid >> 5;
  const int lane = tid & 31;
  const int m    = lane & 15;   // edge row within 16-edge tile
  const int half = lane >> 4;   // 0: lanes 0-15, 1: lanes 16-31

  // ---- stage B = reshape(e2W,[1024,32]) (+ e2b as k-step 32) into LDS, f16,
  //      in this kernel's contiguous-k fragment order (same k order as A): lane n<16 holds B[k=j, col=n] (j=0..15),
  //      lane n>=16 holds B[k=16+j, col=n-16].
  const float* W2 = e2W + layer * (32 * 1024);
  const float* b2 = e2b + layer * 1024;
  for (int idx = tid; idx < SMEM_B_HALFS; idx += 128) {
    int j  = idx & 15;
    int ln = (idx >> 4) & 31;
    int nt = (idx >> 9) & 1;
    int t  = idx >> 10;
    int kk  = (ln & 16) ? (j + 16) : j;       // K within 32-wide step (=c index)
    int col = nt * 16 + (ln & 15);            // output channel o
    float v = (t < 32) ? W2[t * 1024 + kk * 32 + col] : b2[kk * 32 + col];
    Bl[idx] = (_Float16)v;
  }
  float* e1s  = (float*)(smem + OFF_E1W);
  float* e1bs = (float*)(smem + OFF_E1B);
  for (int idx = tid; idx < 256; idx += 128) e1s[idx] = e1W[layer * 256 + idx];
  if (tid < 32) e1bs[tid] = e1b[layer * 32 + tid];
  __syncthreads();

  float* hidW = (float*)(smem + OFF_HID) + wave * (16 * 34);
  int*   dstW = (int*)(smem + OFF_DST) + wave * 16;
  const int* srcI = edge_index;
  const int* dstI = edge_index + N_EDGES;

  // 128 blocks x 4 waves = 512 waves; 8192 tiles; exactly 16 tiles per wave
  // => uniform trip count, in-loop __syncthreads is safe, EXEC always all-1s.
  const int gw = (blockIdx.x * 128 + tid) >> 5;
  for (int tile = gw; tile < N_EDGES / 16; tile += 512) {
    int e = tile * 16 + m;
    unsigned s = (unsigned)srcI[e];
    if (half == 0) dstW[m] = dstI[e];

    // hidden = relu(edge_attr @ e1W + e1b); this lane does 16 units of row m
    const float* ea = edge_attr + e * 8;
    float a0=ea[0],a1=ea[1],a2=ea[2],a3=ea[3],a4=ea[4],a5=ea[5],a6=ea[6],a7=ea[7];
    #pragma unroll
    for (int uu = 0; uu < 16; ++uu) {
      int u = half * 16 + uu;
      float acc = e1bs[u];
      acc += a0 * e1s[0*32+u]; acc += a1 * e1s[1*32+u];
      acc += a2 * e1s[2*32+u]; acc += a3 * e1s[3*32+u];
      acc += a4 * e1s[4*32+u]; acc += a5 * e1s[5*32+u];
      acc += a6 * e1s[6*32+u]; acc += a7 * e1s[7*32+u];
      hidW[m * 34 + u] = fmaxf(acc, 0.0f);
    }
    if (half == 0) hidW[m * 34 + 32] = 1.0f;   // bias k-step scale

    // h[src] as A-fragment (contiguous k order, same as the B fragments):
    // lane<16 holds K = {0..15}; lane>=16 holds K = {16..31}
    const float* hr = hsrc + (size_t)s * 32u;
    v16h a_hs;
    #pragma unroll
    for (int j = 0; j < 8; ++j) a_hs[j]     = (_Float16)hr[half * 16 + j];
    #pragma unroll
    for (int j = 0; j < 8; ++j) a_hs[8 + j] = (_Float16)hr[half * 16 + 8 + j];
    __syncthreads();

    // GEMM: A(step t) = hidden[:,t] (row scalar) * hs  -> 2 WMMA per step.
    // Software pipeline: a/b fragments for step t+1 are produced while the
    // WMMAs for step t are in flight (distinct regs -> no WAR hazard nops,
    // ds_load latency hidden behind the matrix ops).
    v8f acc0 = {}; v8f acc1 = {};
    v16h a_cur = a_hs * (_Float16)hidW[m * 34 + 0];
    v16h b0 = *(const v16h*)(Bl + ((0 * 2 + 0) * 32 + lane) * 16);
    v16h b1 = *(const v16h*)(Bl + ((0 * 2 + 1) * 32 + lane) * 16);
    for (int t = 0; t < 32; ++t) {
      v16h a_nxt = a_hs * (_Float16)hidW[m * 34 + t + 1];
      v16h b0n = *(const v16h*)(Bl + (((t + 1) * 2 + 0) * 32 + lane) * 16);
      v16h b1n = *(const v16h*)(Bl + (((t + 1) * 2 + 1) * 32 + lane) * 16);
      acc0 = __builtin_amdgcn_wmma_f32_16x16x32_f16(false, a_cur, false, b0,
                                                    (short)0, acc0, false, false);
      acc1 = __builtin_amdgcn_wmma_f32_16x16x32_f16(false, a_cur, false, b1,
                                                    (short)0, acc1, false, false);
      a_cur = a_nxt; b0 = b0n; b1 = b1n;
    }
    acc0 = __builtin_amdgcn_wmma_f32_16x16x32_f16(false, a_cur, false, b0,
                                                  (short)0, acc0, false, false);
    acc1 = __builtin_amdgcn_wmma_f32_16x16x32_f16(false, a_cur, false, b1,
                                                  (short)0, acc1, false, false);

    // D layout: VGPR r, lane<16 -> row r col lane; lane>=16 -> row 8+r col lane-16
    int rb = half * 8;
    #pragma unroll
    for (int r = 0; r < 8; ++r) {
      unsigned d = (unsigned)dstW[rb + r];
      float* ap = agg + (size_t)d * 32u;
      atomicAdd(ap + m,      acc0[r]);
      atomicAdd(ap + 16 + m, acc1[r]);
    }
  }
}

// --------------------------- node / graph kernels --------------------------

// h_new = agg/deg + h@root + cbias  (written in place over agg); also graph sums
__global__ void node_update_kernel(const float* __restrict__ h, float* agg,
    const float* __restrict__ root, const float* __restrict__ cbias,
    const float* __restrict__ deg, const int* __restrict__ batch,
    float* __restrict__ gsum, int layer)
{
  int n = blockIdx.x * blockDim.x + threadIdx.x;
  if (n >= N_NODES) return;
  const float* rt = root + layer * 1024;
  const float* cb = cbias + layer * 32;
  float hv[32];
  #pragma unroll
  for (int k = 0; k < 32; ++k) hv[k] = h[(size_t)n * 32 + k];
  float dinv = 1.0f / fmaxf(deg[n], 1.0f);
  unsigned g = (unsigned)batch[n];
  for (int c = 0; c < 32; ++c) {
    float acc = cb[c] + agg[(size_t)n * 32 + c] * dinv;
    #pragma unroll
    for (int k = 0; k < 32; ++k) acc += hv[k] * rt[k * 32 + c];
    agg[(size_t)n * 32 + c] = acc;
    atomicAdd(gsum + g * 32u + c, acc);
  }
}

__global__ void gn_var_kernel(const float* __restrict__ hn,
    const float* __restrict__ gsum, const float* __restrict__ gcnt,
    const float* __restrict__ gn_a, const int* __restrict__ batch,
    float* __restrict__ gvar, int layer)
{
  int n = blockIdx.x * blockDim.x + threadIdx.x;
  if (n >= N_NODES) return;
  unsigned g = (unsigned)batch[n];
  float cinv = 1.0f / fmaxf(gcnt[g], 1.0f);
  for (int c = 0; c < 32; ++c) {
    float mean = gsum[g * 32u + c] * cinv;
    float o = hn[(size_t)n * 32 + c] - gn_a[layer * 32 + c] * mean;
    atomicAdd(gvar + g * 32u + c, o * o);
  }
}

__global__ void gn_apply_kernel(const float* __restrict__ hn,
    const float* __restrict__ gsum, const float* __restrict__ gvar,
    const float* __restrict__ gcnt, const float* __restrict__ gn_w,
    const float* __restrict__ gn_b, const float* __restrict__ gn_a,
    const int* __restrict__ batch, float* __restrict__ hout, int layer)
{
  int n = blockIdx.x * blockDim.x + threadIdx.x;
  if (n >= N_NODES) return;
  unsigned g = (unsigned)batch[n];
  float cinv = 1.0f / fmaxf(gcnt[g], 1.0f);
  for (int c = 0; c < 32; ++c) {
    float mean = gsum[g * 32u + c] * cinv;
    float o = hn[(size_t)n * 32 + c] - gn_a[layer * 32 + c] * mean;
    float var = gvar[g * 32u + c] * cinv;
    float v = gn_w[layer * 32 + c] * o * rsqrtf(var + EPSV) + gn_b[layer * 32 + c];
    hout[(size_t)n * 32 + c] = fmaxf(v, 0.0f);
  }
}

__global__ void pool_kernel(const float* __restrict__ h,
    const int* __restrict__ batch, float* __restrict__ pooled)
{
  int n = blockIdx.x * blockDim.x + threadIdx.x;
  if (n >= N_NODES) return;
  unsigned g = (unsigned)batch[n];
  for (int c = 0; c < 32; ++c)
    atomicAdd(pooled + g * 32u + c, h[(size_t)n * 32 + c]);
}

__global__ void head_kernel(const float* __restrict__ pooled,
    const float* __restrict__ gcnt,
    const float* __restrict__ l1W, const float* __restrict__ l1b,
    const float* __restrict__ ln1g, const float* __restrict__ ln1b,
    const float* __restrict__ l2W, const float* __restrict__ l2b,
    const float* __restrict__ ln2g, const float* __restrict__ ln2b,
    const float* __restrict__ outW, const float* __restrict__ outb,
    float* __restrict__ out)
{
  int g = blockIdx.x * blockDim.x + threadIdx.x;
  if (g >= N_GRAPHS) return;
  float cinv = 1.0f / fmaxf(gcnt[g], 1.0f);
  float pm[32];
  for (int c = 0; c < 32; ++c) pm[c] = pooled[g * 32 + c] * cinv;

  float t1[64];
  for (int u = 0; u < 64; ++u) {
    float acc = l1b[u];
    for (int c = 0; c < 32; ++c) acc += pm[c] * l1W[c * 64 + u];
    t1[u] = acc;
  }
  float mu = 0.f; for (int u = 0; u < 64; ++u) mu += t1[u]; mu *= (1.0f / 64.0f);
  float var = 0.f;
  for (int u = 0; u < 64; ++u) { float d = t1[u] - mu; var += d * d; }
  var *= (1.0f / 64.0f);
  float rs = rsqrtf(var + EPSV);
  for (int u = 0; u < 64; ++u)
    t1[u] = fmaxf(ln1g[u] * (t1[u] - mu) * rs + ln1b[u], 0.0f);

  float t2[32];
  for (int v = 0; v < 32; ++v) {
    float acc = l2b[v];
    for (int u = 0; u < 64; ++u) acc += t1[u] * l2W[u * 32 + v];
    t2[v] = acc;
  }
  mu = 0.f; for (int v = 0; v < 32; ++v) mu += t2[v]; mu *= (1.0f / 32.0f);
  var = 0.f;
  for (int v = 0; v < 32; ++v) { float d = t2[v] - mu; var += d * d; }
  var *= (1.0f / 32.0f);
  rs = rsqrtf(var + EPSV);
  for (int v = 0; v < 32; ++v)
    t2[v] = fmaxf(ln2g[v] * (t2[v] - mu) * rs + ln2b[v], 0.0f);

  float o = outb[0];
  for (int v = 0; v < 32; ++v) o += t2[v] * outW[v];
  out[g] = o;
  for (int v = 0; v < 32; ++v) out[N_GRAPHS + g * 32 + v] = t2[v];
}

// ------------------------------ launch -------------------------------------

extern "C" void kernel_launch(void* const* d_in, const int* in_sizes, int n_in,
                              void* d_out, int out_size, void* d_ws, size_t ws_size,
                              hipStream_t stream)
{
  (void)in_sizes; (void)n_in; (void)out_size; (void)ws_size;
  const float* x    = (const float*)d_in[0];
  const float* ea   = (const float*)d_in[1];
  const float* e1W  = (const float*)d_in[2];
  const float* e1b  = (const float*)d_in[3];
  const float* e2W  = (const float*)d_in[4];
  const float* e2b  = (const float*)d_in[5];
  const float* root = (const float*)d_in[6];
  const float* cbias= (const float*)d_in[7];
  const float* gn_w = (const float*)d_in[8];
  const float* gn_b = (const float*)d_in[9];
  const float* gn_a = (const float*)d_in[10];
  const float* l1W  = (const float*)d_in[11];
  const float* l1b  = (const float*)d_in[12];
  const float* ln1g = (const float*)d_in[13];
  const float* ln1b = (const float*)d_in[14];
  const float* l2W  = (const float*)d_in[15];
  const float* l2b  = (const float*)d_in[16];
  const float* ln2g = (const float*)d_in[17];
  const float* ln2b = (const float*)d_in[18];
  const float* outW = (const float*)d_in[19];
  const float* outb = (const float*)d_in[20];
  const int* edge_index = (const int*)d_in[21];
  const int* batch      = (const int*)d_in[22];

  float* agg    = (float*)d_ws;                       // [N,32] msg sums, then h_new
  float* h1     = agg  + (size_t)N_NODES * 32;        // [N,32]
  float* h2     = h1   + (size_t)N_NODES * 32;        // [N,32]
  float* deg    = h2   + (size_t)N_NODES * 32;        // [N]
  float* gcnt   = deg  + N_NODES;                     // [G]
  float* gsum   = gcnt + N_GRAPHS;                    // [G,32]
  float* gvar   = gsum + N_GRAPHS * 32;               // [G,32]
  float* pooled = gvar + N_GRAPHS * 32;               // [G,32]

  auto Z = [&](float* p, int n) {
    zero_kernel<<<(n + 255) / 256, 256, 0, stream>>>(p, n);
  };

  Z(deg, N_NODES); Z(gcnt, N_GRAPHS);
  deg_kernel<<<N_EDGES / 256, 256, 0, stream>>>(edge_index, deg);
  gcnt_kernel<<<N_NODES / 256, 256, 0, stream>>>(batch, gcnt);

  const float* hcur = x;
  float* houts[2] = { h1, h2 };
  for (int l = 0; l < 2; ++l) {
    Z(agg, N_NODES * 32); Z(gsum, N_GRAPHS * 32); Z(gvar, N_GRAPHS * 32);
    edge_wmma_kernel<<<128, 128, SMEM_TOTAL, stream>>>(hcur, ea, e1W, e1b, e2W,
                                                       e2b, edge_index, agg, l);
    node_update_kernel<<<N_NODES / 256, 256, 0, stream>>>(hcur, agg, root, cbias,
                                                          deg, batch, gsum, l);
    gn_var_kernel<<<N_NODES / 256, 256, 0, stream>>>(agg, gsum, gcnt, gn_a,
                                                     batch, gvar, l);
    gn_apply_kernel<<<N_NODES / 256, 256, 0, stream>>>(agg, gsum, gvar, gcnt,
                                                       gn_w, gn_b, gn_a, batch,
                                                       houts[l], l);
    hcur = houts[l];
  }

  Z(pooled, N_GRAPHS * 32);
  pool_kernel<<<N_NODES / 256, 256, 0, stream>>>(h2, batch, pooled);
  head_kernel<<<N_GRAPHS / 256, 256, 0, stream>>>(pooled, gcnt, l1W, l1b, ln1g,
                                                  ln1b, l2W, l2b, ln2g, ln2b,
                                                  outW, outb, (float*)d_out);
}
